// LM_28089086115870
// MI455X (gfx1250) — matrix operand order repaired and checked by reading, not compiled, not run on hardware
//
#include <hip/hip_runtime.h>
#include <hip/hip_bf16.h>
#include <math.h>

#define V_SZ 32000
#define D_SZ 512
#define H_SZ 1024
#define B_SZ 16
#define S_SZ 256
#define SL   254          /* S-2 */
#define NW   8            /* waves per projection workgroup */
#define LDA  2056         /* padded LDS row stride (ushorts) for A tile */

typedef __attribute__((ext_vector_type(16))) __bf16 v16bf;
typedef __attribute__((ext_vector_type(8)))  float  v8f;

union BfFrag { v16bf v; uint4 u[2]; };

__device__ __forceinline__ unsigned short f2bf(float f) {
  union { float f; unsigned int u; } x; x.f = f;
  unsigned int r = x.u + 0x7FFFu + ((x.u >> 16) & 1u);  // RNE
  return (unsigned short)(r >> 16);
}

__device__ __forceinline__ float sigmoidf(float x) {
  return 1.0f / (1.0f + __expf(-x));
}

// A fragment (16x32 bf16, row-major source, row ptr already offset to row m):
// lanes 0-15: elems0-7 = K kb+0..7,  elems8-15 = K kb+16..23
// lanes16-31: elems0-7 = K kb+8..15, elems8-15 = K kb+24..31
__device__ __forceinline__ v16bf load_a(const unsigned short* row, int kb, int kh) {
  BfFrag f;
  const unsigned short* p = row + kb + kh * 8;
  f.u[0] = *(const uint4*)(p);
  f.u[1] = *(const uint4*)(p + 16);
  return f.v;
}

// B fragment (32x16 bf16): lane holds column n = lane&15; lanes 0-15 K=kb+0..7,kb+16..23,
// lanes 16-31 K=kb+8..15,kb+24..31 (same K order as A, along the weight row).
__device__ __forceinline__ v16bf load_b(const unsigned short* col, int kb, int kh) {
  BfFrag f;
  const unsigned short* p = col + kb + kh * 8;
  f.u[0] = *(const uint4*)(p);
  f.u[1] = *(const uint4*)(p + 16);
  return f.v;
}

// ---------------- prep kernels ----------------

__global__ void k_f32_to_bf16(const float* __restrict__ in,
                              unsigned short* __restrict__ out, int n) {
  for (int i = blockIdx.x * blockDim.x + threadIdx.x; i < n;
       i += gridDim.x * blockDim.x)
    out[i] = f2bf(in[i]);
}

__global__ void k_add_bias(const float* __restrict__ a, const float* __restrict__ b,
                           float* __restrict__ o, int n) {
  int i = blockIdx.x * blockDim.x + threadIdx.x;
  if (i < n) o[i] = a[i] + b[i];
}

__global__ void k_embed(const int* __restrict__ text, const float* __restrict__ emb,
                        unsigned short* __restrict__ xs) {
  int t = blockIdx.x / B_SZ;
  int b = blockIdx.x % B_SZ;
  int tok = text[b * S_SZ + t];
  const float* src = emb + (size_t)tok * D_SZ;
  unsigned short* dst = xs + ((size_t)t * B_SZ + b) * D_SZ;
  for (int d = threadIdx.x; d < D_SZ; d += blockDim.x) dst[d] = f2bf(src[d]);
}

__global__ void k_zero_state(unsigned short* __restrict__ hstate,
                             float* __restrict__ cstate) {
  int i = blockIdx.x * blockDim.x + threadIdx.x;
  if (i < 2 * 2 * B_SZ * H_SZ) hstate[i] = 0;      // both dirs, both parities
  if (i < 2 * B_SZ * H_SZ)     cstate[i] = 0.0f;   // both dirs
}

// ---------------- LSTM step (one launch per time step, fw+bw in grid.y) ---------

__global__ void __launch_bounds__(128)
k_lstm_step(int tt,
            const unsigned short* __restrict__ xs,
            const unsigned short* __restrict__ wih_fw,
            const unsigned short* __restrict__ whh_fw,
            const unsigned short* __restrict__ wih_bw,
            const unsigned short* __restrict__ whh_bw,
            const float* __restrict__ b_fw, const float* __restrict__ b_bw,
            unsigned short* __restrict__ hstate, float* __restrict__ cstate,
            unsigned short* __restrict__ fw_hs, unsigned short* __restrict__ bw_hs) {
  const int hb   = blockIdx.x;        // hidden tile 0..63
  const int dir  = blockIdx.y;        // 0 fw, 1 bw
  const int q    = threadIdx.x >> 5;  // gate quadrant (i,f,g,o)
  const int lane = threadIdx.x & 31;
  const int kh   = lane >> 4;
  const int nn   = lane & 15;

  const int t = dir ? (S_SZ - 1 - tt) : tt;
  const unsigned short* wih  = dir ? wih_bw : wih_fw;
  const unsigned short* whh  = dir ? whh_bw : whh_fw;
  const float*          bias = dir ? b_bw   : b_fw;
  const unsigned short* x     = xs + (size_t)t * B_SZ * D_SZ;
  const unsigned short* hprev = hstate + (dir * 2 + (tt & 1)) * (B_SZ * H_SZ);
  unsigned short*       hnext = hstate + (dir * 2 + ((tt + 1) & 1)) * (B_SZ * H_SZ);
  float*                cs    = cstate + dir * (B_SZ * H_SZ);

  __shared__ float lg[4][16][16];

  const int wrow = q * H_SZ + hb * 16 + nn;          // weight row = gate column
  const unsigned short* arow    = x + (size_t)nn * D_SZ;
  const unsigned short* bcol_ih = wih + (size_t)wrow * D_SZ;
  const unsigned short* hrow    = hprev + (size_t)nn * H_SZ;
  const unsigned short* bcol_hh = whh + (size_t)wrow * H_SZ;

  v8f acc = {0.f, 0.f, 0.f, 0.f, 0.f, 0.f, 0.f, 0.f};
  for (int kb = 0; kb < D_SZ; kb += 32) {
    v16bf a     = load_a(arow, kb, kh);
    v16bf bfrag = load_b(bcol_ih, kb, kh);
    acc = __builtin_amdgcn_wmma_f32_16x16x32_bf16(false, a, false, bfrag,
                                                  (short)0, acc, false, false);
  }
  for (int kb = 0; kb < H_SZ; kb += 32) {
    v16bf a     = load_a(hrow, kb, kh);
    v16bf bfrag = load_b(bcol_hh, kb, kh);
    acc = __builtin_amdgcn_wmma_f32_16x16x32_bf16(false, a, false, bfrag,
                                                  (short)0, acc, false, false);
  }
  const float bq = bias[wrow];
#pragma unroll
  for (int d = 0; d < 8; ++d) lg[q][d + kh * 8][nn] = acc[d] + bq;
  __syncthreads();

  for (int e = threadIdx.x; e < 256; e += 128) {
    const int bm = e >> 4, col = e & 15;
    const float iv = sigmoidf(lg[0][bm][col]);
    const float fv = sigmoidf(lg[1][bm][col]);
    const float gv = tanhf(lg[2][bm][col]);
    const float ov = sigmoidf(lg[3][bm][col]);
    const int hidx = hb * 16 + col;
    const float cold = cs[bm * H_SZ + hidx];
    const float cnew = fv * cold + iv * gv;
    const float hv   = ov * tanhf(cnew);
    cs[bm * H_SZ + hidx] = cnew;
    const unsigned short hbv = f2bf(hv);
    hnext[bm * H_SZ + hidx] = hbv;
    unsigned short* hs = dir ? bw_hs : fw_hs;
    hs[((size_t)t * B_SZ + bm) * H_SZ + hidx] = hbv;
  }
}

// ------- fused projection GEMM + online log-softmax + target NLL ---------------

__global__ void __launch_bounds__(256)
k_proj_nll(const unsigned short* __restrict__ fw_hs,
           const unsigned short* __restrict__ bw_hs,
           const unsigned short* __restrict__ wc, const float* __restrict__ bc,
           const int* __restrict__ text, const float* __restrict__ mask,
           float* __restrict__ nll) {
  extern __shared__ unsigned char smem_raw[];
  unsigned short (*lds_a)[LDA] = (unsigned short(*)[LDA])smem_raw;
  float* red   = (float*)(smem_raw + (size_t)16 * LDA * 2);
  float* red_m = red;                 // [NW][16]
  float* red_s = red + NW * 16;
  float* red_t = red + 2 * NW * 16;

  const int tile = blockIdx.x;        // 254 row tiles of 16 rows
  const int tid  = threadIdx.x;
  const int wave = tid >> 5;
  const int lane = tid & 31;
  const int kh   = lane >> 4;
  const int nn   = lane & 15;

  // Stage A tile (16 x 2048 bf16) = concat(fw_hs[b][t], bw_hs[b][t+2]) rows.
  for (int idx = tid; idx < 16 * 256; idx += 256) {
    const int row = idx >> 8;
    const int j   = idx & 255;        // uint4 chunk (8 ushorts) within the row
    const int r = tile * 16 + row;
    const int b = r / SL, t = r % SL;
    const unsigned short* src =
        (j < 128) ? fw_hs + ((size_t)t * B_SZ + b) * H_SZ + j * 8
                  : bw_hs + ((size_t)(t + 2) * B_SZ + b) * H_SZ + (j - 128) * 8;
    *(uint4*)(&lds_a[row][j * 8]) = *(const uint4*)src;
  }
  __syncthreads();

  int ysr[8];
#pragma unroll
  for (int d = 0; d < 8; ++d) {
    const int r = tile * 16 + d + kh * 8;
    const int b = r / SL, t = r % SL;
    ysr[d] = text[b * S_SZ + t + 1];
  }

  float mx[8], sm[8], tg[8];
#pragma unroll
  for (int d = 0; d < 8; ++d) { mx[d] = -INFINITY; sm[d] = 0.0f; tg[d] = -INFINITY; }

  const unsigned short* arow = &lds_a[nn][0];   // A row m = nn
  for (int nt = wave; nt < V_SZ / 16; nt += NW) {
    const int c = nt * 16 + nn;                 // vocab column for this lane
    const unsigned short* bcol = wc + (size_t)c * (2 * H_SZ);
    if (nt + NW < V_SZ / 16)                    // hint: next Wc tile -> caches
      __builtin_prefetch(wc + (size_t)(c + NW * 16) * (2 * H_SZ), 0, 1);
    v8f acc = {0.f, 0.f, 0.f, 0.f, 0.f, 0.f, 0.f, 0.f};
#pragma unroll 4
    for (int kb = 0; kb < 2 * H_SZ; kb += 32) {
      v16bf a     = load_a(arow, kb, kh);
      v16bf bfrag = load_b(bcol, kb, kh);
      acc = __builtin_amdgcn_wmma_f32_16x16x32_bf16(false, a, false, bfrag,
                                                    (short)0, acc, false, false);
    }
    const float bcv = bc[c];
#pragma unroll
    for (int d = 0; d < 8; ++d) {
      const float v = acc[d] + bcv;
      if (v > mx[d]) { sm[d] = sm[d] * __expf(mx[d] - v) + 1.0f; mx[d] = v; }
      else           { sm[d] += __expf(v - mx[d]); }
      if (c == ysr[d]) tg[d] = v;
    }
  }

  // reduce (max, sumexp, target) across the 16 lanes of each half-wave
#pragma unroll
  for (int off = 8; off > 0; off >>= 1) {
#pragma unroll
    for (int d = 0; d < 8; ++d) {
      const float mo = __shfl_xor(mx[d], off, 32);
      const float so = __shfl_xor(sm[d], off, 32);
      const float to = __shfl_xor(tg[d], off, 32);
      const float M = fmaxf(mx[d], mo);
      sm[d] = sm[d] * __expf(mx[d] - M) + so * __expf(mo - M);
      mx[d] = M;
      tg[d] = fmaxf(tg[d], to);
    }
  }
  if (nn == 0) {
#pragma unroll
    for (int d = 0; d < 8; ++d) {
      red_m[wave * 16 + d + kh * 8] = mx[d];
      red_s[wave * 16 + d + kh * 8] = sm[d];
      red_t[wave * 16 + d + kh * 8] = tg[d];
    }
  }
  __syncthreads();

  if (tid < 16) {
    float M = red_m[tid], S = red_s[tid], T = red_t[tid];
    for (int w = 1; w < NW; ++w) {
      const float mo = red_m[w * 16 + tid];
      const float so = red_s[w * 16 + tid];
      const float to = red_t[w * 16 + tid];
      const float M2 = fmaxf(M, mo);
      S = S * __expf(M - M2) + so * __expf(mo - M2);
      M = M2;
      T = fmaxf(T, to);
    }
    const int r = tile * 16 + tid;
    const int b = r / SL, t = r % SL;
    const float mv = mask[b * S_SZ + t + 1];
    nll[r] = -(T - M - __logf(S)) * mv;   // -log_softmax at target, masked
  }
}

__global__ void k_reduce_loss(const float* __restrict__ p, float* __restrict__ out,
                              int n) {
  __shared__ float smem[256];
  float acc = 0.0f;
  for (int i = threadIdx.x; i < n; i += 256) acc += p[i];
  smem[threadIdx.x] = acc;
  __syncthreads();
  for (int s = 128; s > 0; s >>= 1) {
    if (threadIdx.x < s) smem[threadIdx.x] += smem[threadIdx.x + s];
    __syncthreads();
  }
  if (threadIdx.x == 0) out[0] = smem[0] / (float)n;
}

// ---------------- host side ----------------

extern "C" void kernel_launch(void* const* d_in, const int* in_sizes, int n_in,
                              void* d_out, int out_size, void* d_ws, size_t ws_size,
                              hipStream_t stream) {
  (void)in_sizes; (void)n_in; (void)out_size; (void)ws_size;
  const int*   text    = (const int*)  d_in[0];
  const float* mask    = (const float*)d_in[1];
  const float* emb     = (const float*)d_in[2];
  const float* Wih_fw  = (const float*)d_in[3];
  const float* Whh_fw  = (const float*)d_in[4];
  const float* bih_fw  = (const float*)d_in[5];
  const float* bhh_fw  = (const float*)d_in[6];
  const float* Wih_bw  = (const float*)d_in[7];
  const float* Whh_bw  = (const float*)d_in[8];
  const float* bih_bw  = (const float*)d_in[9];
  const float* bhh_bw  = (const float*)d_in[10];
  const float* Wc      = (const float*)d_in[11];
  const float* bc      = (const float*)d_in[12];
  float* out = (float*)d_out;

  char* ws = (char*)d_ws;
  const size_t OFF_XS     = 0;                                   // 4,194,304
  const size_t OFF_WIH_FW = OFF_XS     + (size_t)S_SZ*B_SZ*D_SZ*2;
  const size_t OFF_WHH_FW = OFF_WIH_FW + (size_t)4*H_SZ*D_SZ*2;
  const size_t OFF_WIH_BW = OFF_WHH_FW + (size_t)4*H_SZ*H_SZ*2;
  const size_t OFF_WHH_BW = OFF_WIH_BW + (size_t)4*H_SZ*D_SZ*2;
  const size_t OFF_WC     = OFF_WHH_BW + (size_t)4*H_SZ*H_SZ*2;
  const size_t OFF_BFW    = OFF_WC     + (size_t)V_SZ*2*H_SZ*2;
  const size_t OFF_BBW    = OFF_BFW    + (size_t)4*H_SZ*4;
  const size_t OFF_HST    = OFF_BBW    + (size_t)4*H_SZ*4;
  const size_t OFF_CST    = OFF_HST    + (size_t)2*2*B_SZ*H_SZ*2;
  const size_t OFF_FWHS   = OFF_CST    + (size_t)2*B_SZ*H_SZ*4;
  const size_t OFF_BWHS   = OFF_FWHS   + (size_t)S_SZ*B_SZ*H_SZ*2;
  const size_t OFF_NLL    = OFF_BWHS   + (size_t)S_SZ*B_SZ*H_SZ*2;

  unsigned short* xs_bf   = (unsigned short*)(ws + OFF_XS);
  unsigned short* wihfw_b = (unsigned short*)(ws + OFF_WIH_FW);
  unsigned short* whhfw_b = (unsigned short*)(ws + OFF_WHH_FW);
  unsigned short* wihbw_b = (unsigned short*)(ws + OFF_WIH_BW);
  unsigned short* whhbw_b = (unsigned short*)(ws + OFF_WHH_BW);
  unsigned short* wc_b    = (unsigned short*)(ws + OFF_WC);
  float*          b_fw    = (float*)(ws + OFF_BFW);
  float*          b_bw    = (float*)(ws + OFF_BBW);
  unsigned short* hstate  = (unsigned short*)(ws + OFF_HST);
  float*          cstate  = (float*)(ws + OFF_CST);
  unsigned short* fw_hs   = (unsigned short*)(ws + OFF_FWHS);
  unsigned short* bw_hs   = (unsigned short*)(ws + OFF_BWHS);
  float*          nll     = (float*)(ws + OFF_NLL);

  // 1) weight/bias precompute (bf16 conversion + fused biases)
  k_f32_to_bf16<<<4096, 256, 0, stream>>>(Wih_fw, wihfw_b, 4*H_SZ*D_SZ);
  k_f32_to_bf16<<<4096, 256, 0, stream>>>(Whh_fw, whhfw_b, 4*H_SZ*H_SZ);
  k_f32_to_bf16<<<4096, 256, 0, stream>>>(Wih_bw, wihbw_b, 4*H_SZ*D_SZ);
  k_f32_to_bf16<<<4096, 256, 0, stream>>>(Whh_bw, whhbw_b, 4*H_SZ*H_SZ);
  k_f32_to_bf16<<<8192, 256, 0, stream>>>(Wc,     wc_b,    V_SZ*2*H_SZ);
  k_add_bias<<<16, 256, 0, stream>>>(bih_fw, bhh_fw, b_fw, 4*H_SZ);
  k_add_bias<<<16, 256, 0, stream>>>(bih_bw, bhh_bw, b_bw, 4*H_SZ);

  // 2) embedding gather -> bf16 time-major activations
  k_embed<<<S_SZ * B_SZ, 256, 0, stream>>>(text, emb, xs_bf);

  // 3) zero LSTM state
  k_zero_state<<<(2*2*B_SZ*H_SZ + 255)/256, 256, 0, stream>>>(hstate, cstate);

  // 4) recurrent sweep: fw & bw directions per step, ping-pong h buffers
  for (int tt = 0; tt < S_SZ; ++tt) {
    k_lstm_step<<<dim3(H_SZ/16, 2), 128, 0, stream>>>(
        tt, xs_bf, wihfw_b, whhfw_b, wihbw_b, whhbw_b, b_fw, b_bw,
        hstate, cstate, fw_hs, bw_hs);
  }

  // 5) fused projection GEMM + online log-softmax + target NLL
  const size_t proj_lds = (size_t)16 * LDA * 2 + (size_t)3 * NW * 16 * 4;
  k_proj_nll<<<SL * B_SZ / 16, 256, proj_lds, stream>>>(
      fw_hs, bw_hs, wc_b, bc, text, mask, nll);

  // 6) masked mean
  k_reduce_loss<<<1, 256, 0, stream>>>(nll, out, B_SZ * SL);
}
